// GroupedQueryAttention_43499428774488
// MI455X (gfx1250) — matrix operand order repaired and checked by reading, not compiled, not run on hardware
//
#include <hip/hip_runtime.h>
#include <hip/hip_bf16.h>

#define DMODEL 1024
#define NHEAD  16
#define NKV    4
#define HDIM   64
#define LSEQ   2048
#define BATCH  2
#define MROWS  (BATCH * LSEQ)      // 4096
#define NQKV   (DMODEL + 2 * NKV * HDIM)  // 1536

typedef _Float16 half16v __attribute__((ext_vector_type(16)));
typedef _Float16 half8v  __attribute__((ext_vector_type(8)));
typedef _Float16 half4v  __attribute__((ext_vector_type(4)));
typedef _Float16 half2v  __attribute__((ext_vector_type(2)));
typedef float    float8v __attribute__((ext_vector_type(8)));

__device__ __forceinline__ half16v frag_cat(half8v lo, half8v hi) {
  half16v f;
#pragma unroll
  for (int j = 0; j < 8; ++j) { f[j] = lo[j]; f[8 + j] = hi[j]; }
  return f;
}

// ---------------------------------------------------------------------------
// One-time prep: f32 -> f16 convert (vectorized), and W transpose+convert.
// ---------------------------------------------------------------------------
__launch_bounds__(256)
__global__ void cvt_f16(const float* __restrict__ in, _Float16* __restrict__ out, int n4) {
  const int i = blockIdx.x * 256 + threadIdx.x;
  if (i < n4) {
    float4 v = ((const float4*)in)[i];
    half4v h; h[0] = (_Float16)v.x; h[1] = (_Float16)v.y;
    h[2] = (_Float16)v.z; h[3] = (_Float16)v.w;
    ((half4v*)out)[i] = h;
  }
}

// in: (K=1024, N) f32 row-major  ->  out: (N, 1024) f16 row-major
__launch_bounds__(256)
__global__ void transpose_w(const float* __restrict__ in, _Float16* __restrict__ out, int N) {
  const int id = blockIdx.x * 256 + threadIdx.x;  // N*1024 total
  const int n = id >> 10, k = id & 1023;
  out[id] = (_Float16)in[(size_t)k * N + n];
}

// ---------------------------------------------------------------------------
// Register-blocked f16 WMMA GEMM.
//   out[M=4096, N] = A(f16, M x 1024) * Bt(f16, N x 1024)^T + bias
//   block: 256 thr = 8 waves; block tile 64(M) x 128(N); wave tile 32x32;
//   K-step 32 -> 4 v_wmma per wave per step.
// MODE 0: A is plain row-major (xh).  MODE 1: A gathered from ctx (B,H,L,64).
// ---------------------------------------------------------------------------
template <int MODE>
__launch_bounds__(256)
__global__ void gemm_f16(const _Float16* __restrict__ A, const _Float16* __restrict__ Bt,
                         const float* __restrict__ b0, const float* __restrict__ b1,
                         const float* __restrict__ b2, float* __restrict__ out, int N) {
  __shared__ _Float16 As[64][40];    // [m][k]
  __shared__ _Float16 Bs[128][40];   // [n][k]
  const int tid  = threadIdx.x;
  const int wave = tid >> 5, lane = tid & 31;
  const int wr = wave >> 2, wc = wave & 3;               // 2(M) x 4(N) waves
  const int m0 = blockIdx.y * 64, n0 = blockIdx.x * 128;
  const int lm = lane & 15, half = lane >> 4;
  float8v acc00 = {}, acc01 = {}, acc10 = {}, acc11 = {};

  const int arow = tid >> 2, ac8 = (tid & 3) << 3;       // A: 64 x 32
  const int brow = tid >> 1, bc16 = (tid & 1) << 4;      // B: 128 x 32

  for (int kb = 0; kb < DMODEL; kb += 32) {
    const _Float16* ap;
    if (MODE == 0) {
      ap = A + (size_t)(m0 + arow) * DMODEL + kb + ac8;
    } else {
      const int m = m0 + arow;
      const int b = m >> 11, l = m & 2047;               // L = 2048
      const int col = kb + ac8;
      const int h = col >> 6, kk = col & 63;
      ap = A + (((size_t)(b * NHEAD + h) * LSEQ + l) * HDIM + kk);
    }
    *(half8v*)&As[arow][ac8] = *(const half8v*)ap;
    const _Float16* bp = Bt + (size_t)(n0 + brow) * DMODEL + kb + bc16;
    *(half8v*)&Bs[brow][bc16]     = *(const half8v*)bp;
    *(half8v*)&Bs[brow][bc16 + 8] = *(const half8v*)(bp + 8);
    if (kb + 32 < DMODEL) {
      __builtin_prefetch(bp + 32, 0, 1);
      __builtin_prefetch(ap + 32, 0, 1);
    }
    __syncthreads();

    const _Float16* ar0 = &As[wr * 32 + lm][0];
    const _Float16* ar1 = &As[wr * 32 + 16 + lm][0];
    half16v af0 = frag_cat(*(const half8v*)(ar0 + half * 8),
                           *(const half8v*)(ar0 + 16 + half * 8));
    half16v af1 = frag_cat(*(const half8v*)(ar1 + half * 8),
                           *(const half8v*)(ar1 + 16 + half * 8));
    const _Float16* br0 = &Bs[wc * 32 + lm][0];
    const _Float16* br1 = &Bs[wc * 32 + 16 + lm][0];
    half16v bf0 = frag_cat(*(const half8v*)(br0 + half * 8),
                           *(const half8v*)(br0 + 16 + half * 8));
    half16v bf1 = frag_cat(*(const half8v*)(br1 + half * 8),
                           *(const half8v*)(br1 + 16 + half * 8));
    acc00 = __builtin_amdgcn_wmma_f32_16x16x32_f16(false, af0, false, bf0, (short)0, acc00, false, false);
    acc01 = __builtin_amdgcn_wmma_f32_16x16x32_f16(false, af0, false, bf1, (short)0, acc01, false, false);
    acc10 = __builtin_amdgcn_wmma_f32_16x16x32_f16(false, af1, false, bf0, (short)0, acc10, false, false);
    acc11 = __builtin_amdgcn_wmma_f32_16x16x32_f16(false, af1, false, bf1, (short)0, acc11, false, false);
    __syncthreads();
  }

#pragma unroll
  for (int mi = 0; mi < 2; ++mi) {
#pragma unroll
    for (int ni = 0; ni < 2; ++ni) {
      const float8v& acc = (mi == 0) ? (ni == 0 ? acc00 : acc01)
                                     : (ni == 0 ? acc10 : acc11);
      const int n = n0 + wc * 32 + ni * 16 + lm;
      const float bval = (n < DMODEL) ? b0[n]
                        : (n < DMODEL + NKV * HDIM) ? b1[n - DMODEL]
                                                    : b2[n - DMODEL - NKV * HDIM];
      const int mbase = m0 + wr * 32 + mi * 16 + half * 8;
#pragma unroll
      for (int r = 0; r < 8; ++r)
        out[(size_t)(mbase + r) * N + n] = acc[r] + bval;
    }
  }
}

// ---------------------------------------------------------------------------
// RMSNorm + RoPE + layout change. One wave per 64-element head row.
// raw: (B,L,1536) f32: [0,1024) q heads, [1024,1280) k heads, [1280,1536) v.
//   q -> qh (B,H,L,64) f16 (norm+rope)
//   k -> kh (B,KV,L,64) f16 (norm+rope)
//   v -> vT (B,KV,64,L) f16 (transposed)
// ---------------------------------------------------------------------------
__launch_bounds__(256)
__global__ void normrope(const float* __restrict__ raw,
                         const float* __restrict__ qw, const float* __restrict__ kw,
                         _Float16* __restrict__ qh, _Float16* __restrict__ kh,
                         _Float16* __restrict__ vT) {
  const int wave = threadIdx.x >> 5, lane = threadIdx.x & 31;
  const int rid = blockIdx.x * 8 + wave;
  const int RQ = BATCH * NHEAD * LSEQ;   // 65536
  const int RK = BATCH * NKV * LSEQ;     // 16384
  const int i2 = lane * 2;

  if (rid < RQ + RK) {
    const bool isq = rid < RQ;
    const int id = isq ? rid : rid - RQ;
    const int HH = isq ? NHEAD : NKV;
    const int b = id / (HH * LSEQ);
    const int h = (id / LSEQ) % HH;
    const int l = id % LSEQ;
    const float* src = raw + (size_t)(b * LSEQ + l) * NQKV +
                       (isq ? 0 : DMODEL) + h * HDIM + i2;
    float2 x = *(const float2*)src;
    float ss = x.x * x.x + x.y * x.y;
#pragma unroll
    for (int off = 16; off; off >>= 1) ss += __shfl_xor(ss, off, 32);
    const float inv = rsqrtf(ss * (1.0f / HDIM) + 1e-6f);
    const float2 w = *(const float2*)((isq ? qw : kw) + i2);
    const float xe = x.x * inv * w.x;
    const float xo = x.y * inv * w.y;
    const float freq = powf(10000.0f, -(float)i2 * (1.0f / HDIM));
    const float ang = (float)l * freq;
    const float c = cosf(ang), s = sinf(ang);
    half2v o; o[0] = (_Float16)(xe * c - xo * s); o[1] = (_Float16)(xe * s + xo * c);
    _Float16* dst = (isq ? qh : kh) + (((size_t)(b * HH + h) * LSEQ + l) * HDIM + i2);
    *(half2v*)dst = o;
  } else {
    const int id = rid - RQ - RK;
    const int b = id / (NKV * LSEQ);
    const int kv = (id / LSEQ) % NKV;
    const int l = id % LSEQ;
    const float* src = raw + (size_t)(b * LSEQ + l) * NQKV +
                       (DMODEL + NKV * HDIM) + kv * HDIM + i2;
    float2 x = *(const float2*)src;
    _Float16* base = vT + ((size_t)(b * NKV + kv) * HDIM) * LSEQ;
    base[(size_t)(i2 + 0) * LSEQ + l] = (_Float16)x.x;
    base[(size_t)(i2 + 1) * LSEQ + l] = (_Float16)x.y;
  }
}

// ---------------------------------------------------------------------------
// Causal flash attention (transposed formulation), GQA.
// One wave per (b, h, 16-row q tile). Bk = 32.
//   S^T = K * Q^T  (C-tile: M=key, N=q-row=lane)  -> per-lane scalar softmax
//   O^T = V^T * P^T (C-tile: M=d,  N=q-row=lane)  -> contiguous B128 ctx stores
// ---------------------------------------------------------------------------
__launch_bounds__(32)
__global__ void attn(const _Float16* __restrict__ qh, const _Float16* __restrict__ kh,
                     const _Float16* __restrict__ vT, _Float16* __restrict__ ctx) {
  __shared__ _Float16 Pt[16][32];        // [q-row][key-in-block]
  const int lane = threadIdx.x, lm = lane & 15, half = lane >> 4;
  const int QT = LSEQ / 16;
  const int qt = blockIdx.x % QT;
  const int h  = (blockIdx.x / QT) % NHEAD;
  const int b  = blockIdx.x / (QT * NHEAD);
  const int kv = h / (NHEAD / NKV);
  const int qrow = qt * 16 + lm;

  // Q as B-operand (k = d contiguous per q-row)
  const _Float16* qr = qh + ((size_t)(b * NHEAD + h) * LSEQ + qrow) * HDIM;
  const half16v qb0 = frag_cat(*(const half8v*)(qr + half * 8),
                               *(const half8v*)(qr + 16 + half * 8));
  const half16v qb1 = frag_cat(*(const half8v*)(qr + 32 + half * 8),
                               *(const half8v*)(qr + 48 + half * 8));

  float8v o0 = {}, o1 = {}, o2 = {}, o3 = {};
  float mrun = -3.0e38f, lrun = 0.0f;    // per-lane scalars (one q-row per lane)

  const _Float16* kbase = kh + (size_t)(b * NKV + kv) * LSEQ * HDIM;
  const _Float16* vbase = vT + (size_t)(b * NKV + kv) * HDIM * LSEQ;
  const int nkb = (qt * 16 + 15) / 32 + 1;

  for (int kb = 0; kb < nkb; ++kb) {
    const int key0 = kb * 32;
    // K as A-operand: tile0 rows key0+lm, tile1 rows key0+16+lm
    const _Float16* kr0 = kbase + (size_t)(key0 + lm) * HDIM;
    const _Float16* kr1 = kr0 + 16 * HDIM;
    half16v k0a = frag_cat(*(const half8v*)(kr0 + half * 8),
                           *(const half8v*)(kr0 + 16 + half * 8));
    half16v k0b = frag_cat(*(const half8v*)(kr0 + 32 + half * 8),
                           *(const half8v*)(kr0 + 48 + half * 8));
    half16v k1a = frag_cat(*(const half8v*)(kr1 + half * 8),
                           *(const half8v*)(kr1 + 16 + half * 8));
    half16v k1b = frag_cat(*(const half8v*)(kr1 + 32 + half * 8),
                           *(const half8v*)(kr1 + 48 + half * 8));
    float8v s0 = {}, s1 = {};
    s0 = __builtin_amdgcn_wmma_f32_16x16x32_f16(false, k0a, false, qb0, (short)0, s0, false, false);
    s0 = __builtin_amdgcn_wmma_f32_16x16x32_f16(false, k0b, false, qb1, (short)0, s0, false, false);
    s1 = __builtin_amdgcn_wmma_f32_16x16x32_f16(false, k1a, false, qb0, (short)0, s1, false, false);
    s1 = __builtin_amdgcn_wmma_f32_16x16x32_f16(false, k1b, false, qb1, (short)0, s1, false, false);

    // mask + per-lane softmax (keys of this lane's q-row live in s0/s1 regs)
    float tm = -3.0e38f;
#pragma unroll
    for (int r = 0; r < 8; ++r) {
      const int k0i = key0 + half * 8 + r;
      float a = s0[r] * 0.125f;                   // 1/sqrt(64)
      float c = s1[r] * 0.125f;
      if (k0i > qrow)      a = -1.0e30f;
      if (k0i + 16 > qrow) c = -1.0e30f;
      s0[r] = a; s1[r] = c;
      tm = fmaxf(tm, fmaxf(a, c));
    }
    tm = fmaxf(tm, __shfl_xor(tm, 16, 32));       // combine the two lane halves
    const float nm = fmaxf(mrun, tm);
    const float sc = expf(mrun - nm);
    float rs = 0.0f;
    float e0[8], e1[8];
#pragma unroll
    for (int r = 0; r < 8; ++r) {
      e0[r] = expf(s0[r] - nm);
      e1[r] = expf(s1[r] - nm);
      rs += e0[r] + e1[r];
    }
    rs += __shfl_xor(rs, 16, 32);
    lrun = lrun * sc + rs;
    mrun = nm;
#pragma unroll
    for (int r = 0; r < 8; ++r) { o0[r] *= sc; o1[r] *= sc; o2[r] *= sc; o3[r] *= sc; }

    // P^T (C layout) -> B layout via LDS [qrow][key]
#pragma unroll
    for (int r = 0; r < 8; ++r) {
      Pt[lm][half * 8 + r]      = (_Float16)e0[r];
      Pt[lm][16 + half * 8 + r] = (_Float16)e1[r];
    }
    __syncthreads();
    half16v pb = frag_cat(*(const half8v*)&Pt[lm][half * 8],
                          *(const half8v*)&Pt[lm][16 + half * 8]);
    // V^T as A-operand: rows d = t*16+lm, k = key contiguous
#pragma unroll
    for (int t = 0; t < 4; ++t) {
      const _Float16* vr = vbase + (size_t)(t * 16 + lm) * LSEQ + key0;
      half16v va = frag_cat(*(const half8v*)(vr + half * 8),
                            *(const half8v*)(vr + 16 + half * 8));
      float8v* op = (t == 0) ? &o0 : (t == 1) ? &o1 : (t == 2) ? &o2 : &o3;
      *op = __builtin_amdgcn_wmma_f32_16x16x32_f16(false, va, false, pb, (short)0, *op, false, false);
    }
    __syncthreads();
  }

  // epilogue: per-lane scalar normalize, contiguous B128 stores
  const float inv = 1.0f / lrun;
  _Float16* crow = ctx + ((size_t)(b * NHEAD + h) * LSEQ + qrow) * HDIM;
#pragma unroll
  for (int t = 0; t < 4; ++t) {
    const float8v& o = (t == 0) ? o0 : (t == 1) ? o1 : (t == 2) ? o2 : o3;
    half8v st;
#pragma unroll
    for (int r = 0; r < 8; ++r) st[r] = (_Float16)(o[r] * inv);
    *(half8v*)(crow + t * 16 + half * 8) = st;
  }
}

// ---------------------------------------------------------------------------
extern "C" void kernel_launch(void* const* d_in, const int* in_sizes, int n_in,
                              void* d_out, int out_size, void* d_ws, size_t ws_size,
                              hipStream_t stream) {
  (void)in_sizes; (void)n_in; (void)out_size; (void)ws_size;
  const float* x   = (const float*)d_in[0];
  const float* Wq  = (const float*)d_in[1];
  const float* bq  = (const float*)d_in[2];
  const float* Wk  = (const float*)d_in[3];
  const float* bk  = (const float*)d_in[4];
  const float* Wv  = (const float*)d_in[5];
  const float* bv  = (const float*)d_in[6];
  const float* Wo  = (const float*)d_in[7];
  const float* bo  = (const float*)d_in[8];
  const float* qnw = (const float*)d_in[9];
  const float* knw = (const float*)d_in[10];

  _Float16* xh    = (_Float16*)d_ws;                         // 4096*1024 f16
  _Float16* wqkvT = xh + (size_t)MROWS * DMODEL;             // 1536*1024 f16
  _Float16* woT   = wqkvT + (size_t)NQKV * DMODEL;           // 1024*1024 f16
  float*    raw   = (float*)(woT + (size_t)DMODEL * DMODEL); // 4096*1536 f32
  _Float16* qh    = (_Float16*)(raw + (size_t)MROWS * NQKV);
  _Float16* kh    = qh + (size_t)BATCH * NHEAD * LSEQ * HDIM;
  _Float16* vT    = kh + (size_t)BATCH * NKV * LSEQ * HDIM;
  _Float16* ctx   = vT + (size_t)BATCH * NKV * HDIM * LSEQ;

  const dim3 blk(256);

  cvt_f16<<<(MROWS * DMODEL / 4 + 255) / 256, blk, 0, stream>>>(x, xh, MROWS * DMODEL / 4);
  transpose_w<<<DMODEL * DMODEL / 256, blk, 0, stream>>>(Wq, wqkvT, DMODEL);
  transpose_w<<<(NKV * HDIM) * DMODEL / 256, blk, 0, stream>>>(Wk, wqkvT + (size_t)DMODEL * DMODEL, NKV * HDIM);
  transpose_w<<<(NKV * HDIM) * DMODEL / 256, blk, 0, stream>>>(Wv, wqkvT + (size_t)(DMODEL + NKV * HDIM) * DMODEL, NKV * HDIM);
  transpose_w<<<DMODEL * DMODEL / 256, blk, 0, stream>>>(Wo, woT, DMODEL);

  gemm_f16<0><<<dim3(NQKV / 128, MROWS / 64), blk, 0, stream>>>(
      xh, wqkvT, bq, bk, bv, raw, NQKV);

  const int rows = BATCH * LSEQ * (NHEAD + NKV + NKV);       // 98304
  normrope<<<rows / 8, blk, 0, stream>>>(raw, qnw, knw, qh, kh, vT);

  attn<<<BATCH * NHEAD * (LSEQ / 16), 32, 0, stream>>>(qh, kh, vT, ctx);

  gemm_f16<1><<<dim3(DMODEL / 128, MROWS / 64), blk, 0, stream>>>(
      ctx, woT, bo, bo, bo, (float*)d_out, DMODEL);
}
